// AxialAttention_42975442764618
// MI455X (gfx1250) — matrix operand order repaired and checked by reading, not compiled, not run on hardware
//
#include <hip/hip_runtime.h>

// Axial attention: 8192 independent problems, L=32, c=64, fp32 in/out.
// One wave32 per problem; bf16 WMMA (16x16x32) for QK^T and P*V.

typedef __attribute__((ext_vector_type(16))) __bf16 v16bf;
typedef __attribute__((ext_vector_type(8)))  __bf16 v8bf;
typedef __attribute__((ext_vector_type(2)))  __bf16 v2bf;
typedef __attribute__((ext_vector_type(8)))  float  v8f;

#define LLEN 32           // axial length (H)
#define CDIM 64           // head dim
#define ROWSTRIDE 2048    // floats between consecutive H rows (= W*C)
#define PROBSTRIDE 65536  // floats between consecutive (b,h,t) blocks (= H*W*C)
#define WAVES 4

// Native f32 -> bf16 (lowers to v_cvt_pk_bf16_f32 on gfx1250).
static __device__ __forceinline__ __bf16 cvt_bf16(float f) {
    return (__bf16)f;
}

// Packed pair convert: prefer the hw packed builtin when declared.
static __device__ __forceinline__ v2bf cvt2_bf16(float a, float b) {
#if __has_builtin(__builtin_amdgcn_cvt_pk_bf16_f32)
    return __builtin_amdgcn_cvt_pk_bf16_f32(a, b);
#else
    v2bf r;
    r[0] = (__bf16)a;
    r[1] = (__bf16)b;
    return r;
#endif
}

// Load 8 floats from p0 and 8 floats from p1 (each 16B-aligned), scale, cvt -> v16bf.
static __device__ __forceinline__ v16bf load16_cvt(const float* p0, const float* p1, float s) {
    float b[16];
    *(float4*)&b[0]  = *(const float4*)(p0);
    *(float4*)&b[4]  = *(const float4*)(p0 + 4);
    *(float4*)&b[8]  = *(const float4*)(p1);
    *(float4*)&b[12] = *(const float4*)(p1 + 4);
    v16bf r;
#pragma unroll
    for (int i = 0; i < 8; ++i) {
        v2bf p = cvt2_bf16(b[2 * i] * s, b[2 * i + 1] * s);
        r[2 * i]     = p[0];
        r[2 * i + 1] = p[1];
    }
    return r;
}

static __device__ __forceinline__ v16bf cat8(v8bf lo, v8bf hi) {
    return __builtin_shufflevector(lo, hi, 0, 1, 2, 3, 4, 5, 6, 7,
                                           8, 9, 10, 11, 12, 13, 14, 15);
}

__global__ __launch_bounds__(WAVES * 32)
void AxialAttention_42975442764618_kernel(const float* __restrict__ q,
                                          const float* __restrict__ k,
                                          const float* __restrict__ v,
                                          float* __restrict__ out, int nProb) {
    __shared__ __align__(16) __bf16 sVT[WAVES][CDIM * LLEN]; // V transposed: [c][m], bf16
    __shared__ __align__(16) float  sS [WAVES][LLEN * LLEN]; // scores, f32
    __shared__ __align__(16) __bf16 sP [WAVES][LLEN * LLEN]; // softmax probs, bf16

    const int tid  = threadIdx.x;
    const int wave = tid >> 5;
    const int lane = tid & 31;
    int p = blockIdx.x * WAVES + wave;
    if (p >= nProb) p = nProb - 1;   // clamp (keeps __syncthreads uniform)

    const int bht = p >> 5;          // (b*8+h)*16+t
    const int w   = p & 31;          // W index
    const size_t base = (size_t)bht * PROBSTRIDE + (size_t)w * CDIM;

    const float* qb = q + base;
    const float* kb = k + base;
    const float* vb = v + base;
    float*       ob = out + base;

    __bf16* vT = sVT[wave];
    float*  S  = sS[wave];
    __bf16* P  = sP[wave];

    const int half = lane >> 4;      // lane half selects K sub-range in WMMA layouts
    const int l15  = lane & 15;

    // ---- Phase A: stage V into LDS, transposed, bf16 (coalesced 256B per row) ----
#pragma unroll 4
    for (int m = 0; m < LLEN; ++m) {
        const float2 t = *(const float2*)(vb + (size_t)m * ROWSTRIDE + 2 * lane);
        vT[(2 * lane)     * LLEN + m] = cvt_bf16(t.x);
        vT[(2 * lane + 1) * LLEN + m] = cvt_bf16(t.y);
    }

    // ---- Phase B: S = (Q*scale) K^T via WMMA, directly from global ----
    // B operand (K matrix): element i -> K[16n+l15][32kk + 16*half + i]
    v16bf kop[2][2];
#pragma unroll
    for (int n = 0; n < 2; ++n)
#pragma unroll
        for (int kk = 0; kk < 2; ++kk) {
            const float* src = kb + (size_t)(16 * n + l15) * ROWSTRIDE + 32 * kk + 16 * half;
            kop[n][kk] = load16_cvt(src, src + 8, 1.0f);
        }

#pragma unroll
    for (int r = 0; r < 2; ++r) {
        // A operand (Q): element i -> c = 32kk + 16*half + i
        v16bf qa[2];
#pragma unroll
        for (int kk = 0; kk < 2; ++kk) {
            const float* src = qb + (size_t)(16 * r + l15) * ROWSTRIDE + 32 * kk + 16 * half;
            qa[kk] = load16_cvt(src, src + 8, 0.125f);   // fold 1/sqrt(64) into Q
        }
#pragma unroll
        for (int n = 0; n < 2; ++n) {
            v8f acc = {};
            acc = __builtin_amdgcn_wmma_f32_16x16x32_bf16(false, qa[0], false, kop[n][0],
                                                          (short)0, acc, false, false);
            acc = __builtin_amdgcn_wmma_f32_16x16x32_bf16(false, qa[1], false, kop[n][1],
                                                          (short)0, acc, false, false);
            // D layout: VGPR j -> row 16r + j + 8*half, col 16n + l15
#pragma unroll
            for (int j = 0; j < 8; ++j)
                S[(16 * r + j + 8 * half) * LLEN + 16 * n + l15] = acc[j];
        }
    }

    __syncthreads();

    // ---- Phase C: softmax, one lane per row ----
    float sv[LLEN];
#pragma unroll
    for (int j = 0; j < 8; ++j)
        *(float4*)&sv[4 * j] = *(const float4*)&S[lane * LLEN + 4 * j];
    float mx = sv[0];
#pragma unroll
    for (int j = 1; j < LLEN; ++j) mx = fmaxf(mx, sv[j]);
    float sum = 0.0f;
#pragma unroll
    for (int j = 0; j < LLEN; ++j) { sv[j] = __expf(sv[j] - mx); sum += sv[j]; }
    const float rinv = 1.0f / sum;
#pragma unroll
    for (int j = 0; j < LLEN / 2; ++j) {
        v2bf pr = cvt2_bf16(sv[2 * j] * rinv, sv[2 * j + 1] * rinv);
        *(v2bf*)&P[lane * LLEN + 2 * j] = pr;
    }

    __syncthreads();

    // ---- Phase D: O = P * V via WMMA ----
    // B operand (V): element i -> V[16*half + i][16cn + l15] = vT[(16cn+l15)*32 + 16*half + i]
    v16bf vop[4];
#pragma unroll
    for (int cn = 0; cn < 4; ++cn) {
        const __bf16* src = &vT[(16 * cn + l15) * LLEN + 16 * half];
        vop[cn] = cat8(*(const v8bf*)src, *(const v8bf*)(src + 8));
    }

#pragma unroll
    for (int r = 0; r < 2; ++r) {
        // A operand (P): element i -> m = 16*half + i
        const __bf16* prow = &P[(16 * r + l15) * LLEN];
        v16bf pa = cat8(*(const v8bf*)(prow + 16 * half),
                        *(const v8bf*)(prow + 16 * half + 8));
#pragma unroll
        for (int cn = 0; cn < 4; ++cn) {
            v8f o = {};
            o = __builtin_amdgcn_wmma_f32_16x16x32_bf16(false, pa, false, vop[cn],
                                                        (short)0, o, false, false);
#pragma unroll
            for (int j = 0; j < 8; ++j)
                ob[(size_t)(16 * r + j + 8 * half) * ROWSTRIDE + 16 * cn + l15] = o[j];
        }
    }
}

extern "C" void kernel_launch(void* const* d_in, const int* in_sizes, int n_in,
                              void* d_out, int out_size, void* d_ws, size_t ws_size,
                              hipStream_t stream) {
    const float* q = (const float*)d_in[0];
    const float* k = (const float*)d_in[1];
    const float* v = (const float*)d_in[2];
    float* out = (float*)d_out;

    const int nProb  = in_sizes[0] / (LLEN * CDIM);   // 8192 for the reference shape
    const int blocks = (nProb + WAVES - 1) / WAVES;   // 2048 blocks of 128 threads

    AxialAttention_42975442764618_kernel<<<blocks, WAVES * 32, 0, stream>>>(q, k, v, out, nProb);
}
